// MILPGCNNonRes_45930380264305
// MI455X (gfx1250) — matrix operand order repaired and checked by reading, not compiled, not run on hardware
//
#include <hip/hip_runtime.h>
#include <cstdint>
#include <cstddef>

// ---------------- constants ----------------
constexpr int   HD      = 128;   // hidden width H
constexpr int   DINW    = 16;    // input feature width
constexpr int   DINP    = 32;    // padded input width (K multiple of 32)
constexpr int   NLAYERS = 2;
constexpr int   MAXINT  = 32;
constexpr float EPSV    = 1e-5f;

typedef __attribute__((ext_vector_type(16))) _Float16 v16h;
typedef __attribute__((ext_vector_type(8)))  float    v8f;

// =====================================================================
// Weight pack: W[K,N] f32 row-major -> f16 in WMMA B-fragment order.
//   Wp index (v16h units): (kb*N + n)*2 + hi   holds j=0..15 halves,
//   where k = kb*32 + hi*16 + j.  Lane (n,hi) then loads ONE v16h (32B).
// =====================================================================
__global__ void k_packW(const float* __restrict__ W, int ldw,
                        _Float16* __restrict__ Wp, int K, int N)
{
    int i = blockIdx.x * blockDim.x + threadIdx.x;
    if (i >= K * N) return;
    int k = i / N, n = i - k * N;
    int kb = k >> 5, rem = k & 31, hi = rem >> 4, j = rem & 15;
    Wp[((((long)kb * N + n) * 2 + hi) << 4) + j] = (_Float16)W[(long)k * ldw + n];
}

// =====================================================================
// WMMA GEMM:  C[M,N] (+)= act( A[M,K] @ W[K,N] + biasTerm )
//   One wave computes a 16 x N strip (NT = N/16 accumulators): the A
//   fragment is loaded & converted once per 32-wide k-step and feeds NT
//   back-to-back v_wmma ops (no D->A/B hazard between them).
//   REQUIRES: K,K0 % 32 == 0, N == NT*16, A rows 16B aligned.
//   A split at K0: cols [0,K0) from A0, cols [K0,K) from A1.
//   Template flags: HASBIAS add bias[n]; HASROW scale bias by rowScale[m];
//   ACT relu; ACCUM add into existing C.
// =====================================================================
template <int NT, bool HASBIAS, bool HASROW, bool ACT, bool ACCUM>
__global__ void k_gemm_wmma(const float* __restrict__ A0, int lda0,
                            const float* __restrict__ A1, int lda1, int K0,
                            const _Float16* __restrict__ Wp,
                            const float* __restrict__ bias,
                            const float* __restrict__ rowScale,
                            float* __restrict__ C, int ldc,
                            int M, int K)
{
    const int N    = NT * 16;
    const int wid  = threadIdx.x >> 5;
    const int lane = threadIdx.x & 31;
    const int tilesM = (M + 15) >> 4;
    const int tile = blockIdx.x * (blockDim.x >> 5) + wid;
    if (tile >= tilesM) return;                    // wave-uniform exit
    const int mBase = tile << 4;

    const int row = lane & 15;
    const int hi  = lane >> 4;
    const int m   = mBase + row;                   // A row handled by this lane
    const int mc  = (m < M) ? m : (M - 1);         // clamped row -> unconditional loads

    const float* __restrict__ rowA0 = A0 + (long)mc * lda0;
    const float* __restrict__ rowA1 = A1 + (long)mc * lda1;
    const v16h*  __restrict__ WpV   = reinterpret_cast<const v16h*>(Wp);

    v8f acc[NT];
    #pragma unroll
    for (int nt = 0; nt < NT; ++nt) acc[nt] = (v8f){};

    for (int kk = 0; kk < K; kk += 32) {
        const float* __restrict__ base;            // wave-uniform A0/A1 select
        int kof;
        if (kk < K0) { base = rowA0; kof = kk; }
        else         { base = rowA1; kof = kk - K0; }

        // A fragment: lane holds one contiguous 16-float run k = 16*hi + s (16B aligned)
        const float4* pA  = reinterpret_cast<const float4*>(base + kof + 16 * hi);
        const float4* pB4 = reinterpret_cast<const float4*>(base + kof + 16 * hi + 8);
        float4 q0 = pA[0],  q1 = pA[1];
        float4 q2 = pB4[0], q3 = pB4[1];
        v16h a;
        a[0]=(_Float16)q0.x;  a[1]=(_Float16)q0.y;  a[2]=(_Float16)q0.z;  a[3]=(_Float16)q0.w;
        a[4]=(_Float16)q1.x;  a[5]=(_Float16)q1.y;  a[6]=(_Float16)q1.z;  a[7]=(_Float16)q1.w;
        a[8]=(_Float16)q2.x;  a[9]=(_Float16)q2.y;  a[10]=(_Float16)q2.z; a[11]=(_Float16)q2.w;
        a[12]=(_Float16)q3.x; a[13]=(_Float16)q3.y; a[14]=(_Float16)q3.z; a[15]=(_Float16)q3.w;

        // B fragments: one 32B v16h load per N-tile, then back-to-back WMMAs
        const v16h* __restrict__ wk = WpV + ((long)(kk >> 5) * N) * 2 + hi;
        #pragma unroll
        for (int nt = 0; nt < NT; ++nt) {
            v16h b = wk[(nt * 16 + row) << 1];
            acc[nt] = __builtin_amdgcn_wmma_f32_16x16x32_f16(false, a, false, b,
                                                             (short)0, acc[nt], false, false);
        }
    }

    // ---------------- store (C 16x16 f32: VGPR r -> M=r / r+8 per lane half) ----
    float rs[8];
    if (HASROW) {
        #pragma unroll
        for (int r = 0; r < 8; ++r) {
            int mm = mBase + r + 8 * hi;
            rs[r] = rowScale[(mm < M) ? mm : (M - 1)];
        }
    }
    const bool fullM = (mBase + 16 <= M);
    #pragma unroll
    for (int nt = 0; nt < NT; ++nt) {
        const int n = nt * 16 + row;
        const float bn = HASBIAS ? bias[n] : 0.f;
        float* __restrict__ cp = C + (long)(mBase + 8 * hi) * ldc + n;
        if (fullM) {
            #pragma unroll
            for (int r = 0; r < 8; ++r) {
                float v = acc[nt][r];
                if (ACCUM) v += cp[(long)r * ldc];
                if (HASBIAS) v += HASROW ? rs[r] * bn : bn;
                if (ACT) v = fmaxf(v, 0.f);
                cp[(long)r * ldc] = v;
            }
        } else {
            #pragma unroll
            for (int r = 0; r < 8; ++r) {
                if (mBase + r + 8 * hi < M) {
                    float v = acc[nt][r];
                    if (ACCUM) v += cp[(long)r * ldc];
                    if (HASBIAS) v += HASROW ? rs[r] * bn : bn;
                    if (ACT) v = fmaxf(v, 0.f);
                    cp[(long)r * ldc] = v;
                }
            }
        }
    }
}

// =====================================================================
// Elementwise / helper kernels
// =====================================================================
__global__ void k_zero(float* p, long n)
{
    long i = (long)blockIdx.x * blockDim.x + threadIdx.x;
    long s = (long)gridDim.x * blockDim.x;
    for (; i < n; i += s) p[i] = 0.f;
}

// widen [M,16] -> [M,32], zero padded (so encoder GEMM has K=32)
__global__ void k_pad16to32(const float* __restrict__ X, float* __restrict__ Y, int M)
{
    long n = (long)M * DINP;
    long i = (long)blockIdx.x * blockDim.x + threadIdx.x;
    long s = (long)gridDim.x * blockDim.x;
    for (; i < n; i += s) {
        int c = (int)(i & (DINP - 1));
        long r = i >> 5;
        Y[i] = (c < DINW) ? X[r * DINW + c] : 0.f;
    }
}

__global__ void k_degree(const int* __restrict__ idx, int E, float* __restrict__ deg)
{
    long i = (long)blockIdx.x * blockDim.x + threadIdx.x;
    long s = (long)gridDim.x * blockDim.x;
    for (; i < E; i += s) atomicAdd(&deg[idx[i]], 1.0f);
}

// pre = Ldst[dst] + Rsrc[src] + eattr*edge_w ; h = relu(LN(pre)) ; S[dst] += h
__global__ void k_edge(const float* __restrict__ Ld, const float* __restrict__ Rs,
                       const float* __restrict__ eattr,
                       const int* __restrict__ src_idx, const int* __restrict__ dst_idx,
                       const float* __restrict__ ew, const float* __restrict__ fg,
                       const float* __restrict__ fb, float* __restrict__ S, int E)
{
    __shared__ float red[HD];
    int e = blockIdx.x;
    if (e >= E) return;
    int c = threadIdx.x;
    int s = src_idx[e], d = dst_idx[e];
    float pre = Ld[(long)d * HD + c] + Rs[(long)s * HD + c] + eattr[e] * ew[c];
    red[c] = pre; __syncthreads();
    for (int t = 64; t > 0; t >>= 1) { if (c < t) red[c] += red[c + t]; __syncthreads(); }
    float mean = red[0] * (1.0f / HD); __syncthreads();
    float dv = pre - mean;
    red[c] = dv * dv; __syncthreads();
    for (int t = 64; t > 0; t >>= 1) { if (c < t) red[c] += red[c + t]; __syncthreads(); }
    float var = red[0] * (1.0f / HD);
    float h = dv * rsqrtf(var + EPSV) * fg[c] + fb[c];
    h = fmaxf(h, 0.f);
    atomicAdd(&S[(long)d * HD + c], h);
}

// per-row LayerNorm over 128 cols; mode 0: LN, mode 1: SiLU(LN)
__global__ void k_ln_rows(const float* __restrict__ X, float* __restrict__ Y,
                          const float* __restrict__ g, const float* __restrict__ b,
                          int M, int mode)
{
    __shared__ float red[HD];
    int r = blockIdx.x;
    if (r >= M) return;
    int c = threadIdx.x;
    float x = X[(long)r * HD + c];
    red[c] = x; __syncthreads();
    for (int t = 64; t > 0; t >>= 1) { if (c < t) red[c] += red[c + t]; __syncthreads(); }
    float mean = red[0] * (1.0f / HD); __syncthreads();
    float dv = x - mean;
    red[c] = dv * dv; __syncthreads();
    for (int t = 64; t > 0; t >>= 1) { if (c < t) red[c] += red[c + t]; __syncthreads(); }
    float var = red[0] * (1.0f / HD);
    float v = dv * rsqrtf(var + EPSV) * g[c] + b[c];
    if (mode == 1) v = v * (1.0f / (1.0f + expf(-v)));     // SiLU
    Y[(long)r * HD + c] = v;
}

// time embedding -> time MLP -> per-layer te; writes teI[l] = te_l + oi.b,
// teC[l] = te_l + ov.b  (so te rides the per_out GEMM bias path)
struct TeP {
    const float* tw[NLAYERS]; const float* tb[NLAYERS];
    const float* oib[NLAYERS]; const float* ovb[NLAYERS];
};
__global__ void k_te(const float* __restrict__ ts,
                     const float* __restrict__ b1, const float* __restrict__ b2,
                     const float* __restrict__ w1, const float* __restrict__ w2,
                     TeP p, float* __restrict__ teI, float* __restrict__ teC)
{
    __shared__ float emb[HD];
    __shared__ float h1[64];
    __shared__ float rt[64];
    int c = threadIdx.x;
    float t = ts[0];
    // timestep_embedding(t, 128): [cos(t*f_0..63), sin(t*f_0..63)]
    {
        int j = (c < 64) ? c : c - 64;
        float fr = expf(-9.2103403719761836f * (float)j / 64.0f);
        emb[c] = (c < 64) ? cosf(t * fr) : sinf(t * fr);
    }
    __syncthreads();
    if (c < 64) {                      // hidden = relu(emb @ w1 + b1)   w1:[128,64]
        float s = b1[c];
        for (int k = 0; k < HD; ++k) s += emb[k] * w1[k * 64 + c];
        h1[c] = fmaxf(s, 0.f);
    }
    __syncthreads();
    if (c < 64) {                      // rt = relu(h1 @ w2 + b2)        w2:[64,64]
        float s = b2[c];
        for (int k = 0; k < 64; ++k) s += h1[k] * w2[k * 64 + c];
        rt[c] = fmaxf(s, 0.f);         // relu(temb)
    }
    __syncthreads();
    for (int l = 0; l < NLAYERS; ++l) { // te_l = rt @ time_w + time_b   time_w:[64,128]
        float s = p.tb[l][c];
        for (int k = 0; k < 64; ++k) s += rt[k] * p.tw[l][k * HD + c];
        teI[l * HD + c] = s + p.oib[l][c];
        teC[l * HD + c] = s + p.ovb[l][c];
    }
}

// GroupNorm stats: per group (4 channels), sums over all rows -> gst[2*g],[2*g+1]
__global__ void k_gn_stats(const float* __restrict__ X, int M, float* __restrict__ gst)
{
    __shared__ float s1[256], s2[256];
    int g = blockIdx.x, tid = threadIdx.x;
    float a1 = 0.f, a2 = 0.f;
    for (int r = blockIdx.y * blockDim.x + tid; r < M; r += gridDim.y * blockDim.x) {
        const float* row = X + (long)r * HD + g * 4;
        #pragma unroll
        for (int j = 0; j < 4; ++j) { float v = row[j]; a1 += v; a2 += v * v; }
    }
    s1[tid] = a1; s2[tid] = a2; __syncthreads();
    for (int t = 128; t > 0; t >>= 1) {
        if (tid < t) { s1[tid] += s1[tid + t]; s2[tid] += s2[tid + t]; }
        __syncthreads();
    }
    if (tid == 0) { atomicAdd(&gst[2 * g], s1[0]); atomicAdd(&gst[2 * g + 1], s2[0]); }
}

__global__ void k_gn_apply_relu(const float* __restrict__ X, float* __restrict__ Y,
                                const float* __restrict__ gst,
                                const float* __restrict__ gw, const float* __restrict__ gb,
                                int M)
{
    long n = (long)M * HD;
    float inv = 1.0f / ((float)M * 4.0f);
    long i = (long)blockIdx.x * blockDim.x + threadIdx.x;
    long s = (long)gridDim.x * blockDim.x;
    for (; i < n; i += s) {
        int c = (int)(i & (HD - 1));
        int g = c >> 2;
        float mean = gst[2 * g] * inv;
        float var  = gst[2 * g + 1] * inv - mean * mean;
        float v = (X[i] - mean) * rsqrtf(var + EPSV) * gw[c] + gb[c];
        Y[i] = fmaxf(v, 0.f);
    }
}

// out[r] = dot(X[r,:], w) + wb[0]      (Cout head, N=1)
__global__ void k_rowdot(const float* __restrict__ X, const float* __restrict__ w,
                         const float* __restrict__ wb, float* __restrict__ out, int M)
{
    __shared__ float red[HD];
    int r = blockIdx.x;
    if (r >= M) return;
    int c = threadIdx.x;
    red[c] = X[(long)r * HD + c] * w[c]; __syncthreads();
    for (int t = 64; t > 0; t >>= 1) { if (c < t) red[c] += red[c + t]; __syncthreads(); }
    if (c == 0) out[r] = red[0] + wb[0];
}

// =====================================================================
// Host-side parameter structs (pointers into d_in, jax tree_flatten order)
// =====================================================================
struct MsgP  { const float *edge_w,*fin_b,*fin_w,*fln_b,*fln_g,*left_b,*left_w,*right_w; };
struct OutbP { const float *out1_b,*out1_w,*out2_b,*out2_w,*post_b,*post_g; };
struct PerOutP { const float *b,*ln_b,*ln_g,*w; };
struct LayerP {
    MsgP bC, bI, triC, triI;
    OutbP bC_out, bI_out, tri_out;
    PerOutP oc, oi, ov;
    const float *time_b, *time_w;
};

// pack W then dispatch to the used template instantiations
static inline void launch_gemm(hipStream_t st, _Float16* Wp,
                               const float* A0, int lda0, const float* A1, int lda1, int K0,
                               const float* W, int ldw, const float* bias, const float* rowScale,
                               float* C, int ldc, int M, int N, int K, int act, int accum)
{
    k_packW<<<(K * N + 255) / 256, 256, 0, st>>>(W, ldw, Wp, K, N);
    int tilesM = (M + 15) / 16;
    int blocks = (tilesM + 7) / 8;
    if (N == MAXINT) {                      // head: N=32, bias only
        k_gemm_wmma<2, true, false, false, false><<<blocks, 256, 0, st>>>(
            A0, lda0, A1, lda1, K0, Wp, bias, rowScale, C, ldc, M, K);
        return;
    }
    // N == 128
    if (rowScale) {
        if (accum)
            k_gemm_wmma<8, true, true, false, true><<<blocks, 256, 0, st>>>(
                A0, lda0, A1, lda1, K0, Wp, bias, rowScale, C, ldc, M, K);
        else
            k_gemm_wmma<8, true, true, false, false><<<blocks, 256, 0, st>>>(
                A0, lda0, A1, lda1, K0, Wp, bias, rowScale, C, ldc, M, K);
    } else if (act) {
        k_gemm_wmma<8, true, false, true, false><<<blocks, 256, 0, st>>>(
            A0, lda0, A1, lda1, K0, Wp, bias, rowScale, C, ldc, M, K);
    } else if (bias) {
        k_gemm_wmma<8, true, false, false, false><<<blocks, 256, 0, st>>>(
            A0, lda0, A1, lda1, K0, Wp, bias, rowScale, C, ldc, M, K);
    } else {
        k_gemm_wmma<8, false, false, false, false><<<blocks, 256, 0, st>>>(
            A0, lda0, A1, lda1, K0, Wp, bias, rowScale, C, ldc, M, K);
    }
}

static inline void launch_zero(hipStream_t st, float* p, long n)
{
    int blocks = (int)(((n + 255) / 256) < 4096 ? ((n + 255) / 256) : 4096);
    if (blocks < 1) blocks = 1;
    k_zero<<<blocks, 256, 0, st>>>(p, n);
}

extern "C" void kernel_launch(void* const* d_in, const int* in_sizes, int n_in,
                              void* d_out, int out_size, void* d_ws, size_t ws_size,
                              hipStream_t stream)
{
    auto F = [&](int i) { return (const float*)d_in[i]; };
    int p = 0;

    // ---- params, jax tree_flatten order (dict keys sorted lexicographically) ----
    // Cout: b, g, w, wb
    const float *Cout_b = F(p++), *Cout_g = F(p++), *Cout_w = F(p++), *Cout_wb = F(p++);
    // Iout: b, g, w, wb
    const float *Iout_b = F(p++), *Iout_g = F(p++), *Iout_w = F(p++), *Iout_wb = F(p++);
    // enc_C / enc_I / enc_cons: b1, b2, w1, w2
    const float *encC_b1 = F(p++), *encC_b2 = F(p++), *encC_w1 = F(p++), *encC_w2 = F(p++);
    const float *encI_b1 = F(p++), *encI_b2 = F(p++), *encI_w1 = F(p++), *encI_w2 = F(p++);
    const float *encc_b1 = F(p++), *encc_b2 = F(p++), *encc_w1 = F(p++), *encc_w2 = F(p++);

    LayerP L[NLAYERS];
    for (int l = 0; l < NLAYERS; ++l) {
        LayerP& y = L[l];
        // bC: edge_w,fin_b,fin_w,fln_b,fln_g,left_b,left_w,out1_b,out1_w,out2_b,out2_w,post_b,post_g,right_w
        y.bC.edge_w = F(p++); y.bC.fin_b = F(p++); y.bC.fin_w = F(p++);
        y.bC.fln_b = F(p++);  y.bC.fln_g = F(p++);
        y.bC.left_b = F(p++); y.bC.left_w = F(p++);
        y.bC_out.out1_b = F(p++); y.bC_out.out1_w = F(p++);
        y.bC_out.out2_b = F(p++); y.bC_out.out2_w = F(p++);
        y.bC_out.post_b = F(p++); y.bC_out.post_g = F(p++);
        y.bC.right_w = F(p++);
        // bI: same layout
        y.bI.edge_w = F(p++); y.bI.fin_b = F(p++); y.bI.fin_w = F(p++);
        y.bI.fln_b = F(p++);  y.bI.fln_g = F(p++);
        y.bI.left_b = F(p++); y.bI.left_w = F(p++);
        y.bI_out.out1_b = F(p++); y.bI_out.out1_w = F(p++);
        y.bI_out.out2_b = F(p++); y.bI_out.out2_w = F(p++);
        y.bI_out.post_b = F(p++); y.bI_out.post_g = F(p++);
        y.bI.right_w = F(p++);
        // oc / oi / ov: b, ln_b, ln_g, w
        y.oc.b = F(p++); y.oc.ln_b = F(p++); y.oc.ln_g = F(p++); y.oc.w = F(p++);
        y.oi.b = F(p++); y.oi.ln_b = F(p++); y.oi.ln_g = F(p++); y.oi.w = F(p++);
        y.ov.b = F(p++); y.ov.ln_b = F(p++); y.ov.ln_g = F(p++); y.ov.w = F(p++);
        // time_b, time_w
        y.time_b = F(p++); y.time_w = F(p++);
        // tri: C(msg 8), I(msg 8), out1_b,out1_w,out2_b,out2_w,post_b,post_g
        y.triC.edge_w = F(p++); y.triC.fin_b = F(p++); y.triC.fin_w = F(p++);
        y.triC.fln_b = F(p++);  y.triC.fln_g = F(p++);
        y.triC.left_b = F(p++); y.triC.left_w = F(p++); y.triC.right_w = F(p++);
        y.triI.edge_w = F(p++); y.triI.fin_b = F(p++); y.triI.fin_w = F(p++);
        y.triI.fln_b = F(p++);  y.triI.fln_g = F(p++);
        y.triI.left_b = F(p++); y.triI.left_w = F(p++); y.triI.right_w = F(p++);
        y.tri_out.out1_b = F(p++); y.tri_out.out1_w = F(p++);
        y.tri_out.out2_b = F(p++); y.tri_out.out2_w = F(p++);
        y.tri_out.post_b = F(p++); y.tri_out.post_g = F(p++);
    }
    // time MLP: b1, b2, w1, w2
    const float *time_b1 = F(p++), *time_b2 = F(p++), *time_w1 = F(p++), *time_w2 = F(p++);

    // ---- data arrays (setup_inputs insertion order after params) ----
    const int   ix_xC = p;     const float* x_Cvars = F(p++);
    const int   ix_xI = p;     const float* x_Ivars = F(p++);
    const int   ix_xc = p;     const float* x_cons  = F(p++);
    const int   ix_Ca = p;     const float* C_attr  = F(p++);
    const int   ix_Ia = p;     const float* I_attr  = F(p++);
    const int*  C_idx = (const int*)d_in[p++];
    const int*  I_idx = (const int*)d_in[p++];
    const float* tsteps = F(p++);

    const int N_C    = in_sizes[ix_xC] / DINW;
    const int N_I    = in_sizes[ix_xI] / DINW;
    const int N_CONS = in_sizes[ix_xc] / DINW;
    const int E_C    = in_sizes[ix_Ca];
    const int E_I    = in_sizes[ix_Ia];
    const int* C_src = C_idx;            const int* C_dst = C_idx + E_C;
    const int* I_src = I_idx;            const int* I_dst = I_idx + E_I;

    // ---- workspace layout (floats; every region size is a multiple of 8) ----
    int NMX = N_I > N_C ? N_I : N_C;
    if (N_CONS > NMX) NMX = N_CONS;
    float* w0 = (float*)d_ws;
    size_t off = 0;
    float* bufA = w0 + off; off += (size_t)NMX * HD;
    float* bufB = w0 + off; off += (size_t)NMX * HD;
    float* bufS = w0 + off; off += (size_t)NMX * HD;
    float* bufT = w0 + off; off += (size_t)NMX * HD;
    float* xI   = w0 + off; off += (size_t)N_I * HD;
    float* xC   = w0 + off; off += (size_t)N_C * HD;
    float* xc   = w0 + off; off += (size_t)N_CONS * HD;
    float* degIc = w0 + off; off += (size_t)(N_CONS + 7) & ~(size_t)7;
    float* degCc = w0 + off; off += (size_t)(N_CONS + 7) & ~(size_t)7;
    float* degIv = w0 + off; off += (size_t)(N_I + 7) & ~(size_t)7;
    float* degCv = w0 + off; off += (size_t)(N_C + 7) & ~(size_t)7;
    float* teI   = w0 + off; off += NLAYERS * HD;
    float* teC   = w0 + off; off += NLAYERS * HD;
    float* gst   = w0 + off; off += 64;
    _Float16* Wp = (_Float16*)(w0 + off); off += (2 * HD) * HD / 2;  // 256x128 halves

    float* outI = (float*)d_out;               // [N_I, 32]
    float* outC = outI + (size_t)N_I * MAXINT; // [N_C]

    // ---- degrees (per edge set, both directions) ----
    launch_zero(stream, degIc, N_CONS); launch_zero(stream, degCc, N_CONS);
    launch_zero(stream, degIv, N_I);    launch_zero(stream, degCv, N_C);
    {
        int bl = (E_I + 255) / 256; if (bl > 4096) bl = 4096;
        k_degree<<<bl, 256, 0, stream>>>(I_dst, E_I, degIc);
        k_degree<<<bl, 256, 0, stream>>>(I_src, E_I, degIv);
        bl = (E_C + 255) / 256; if (bl > 4096) bl = 4096;
        k_degree<<<bl, 256, 0, stream>>>(C_dst, E_C, degCc);
        k_degree<<<bl, 256, 0, stream>>>(C_src, E_C, degCv);
    }

    // ---- encoders: pad [N,16]->[N,32], then K=32 GEMM -> relu -> K=128 GEMM ----
    auto encoder = [&](const float* xin, int n,
                       const float* w1, const float* b1,
                       const float* w2, const float* b2, float* target)
    {
        int bl = (int)((((long)n * DINP + 255) / 256) < 4096 ? (((long)n * DINP + 255) / 256) : 4096);
        k_pad16to32<<<bl, 256, 0, stream>>>(xin, bufB, n);
        launch_gemm(stream, Wp, bufB, DINP, bufB, DINP, DINP, w1, HD, b1, nullptr,
                    bufA, HD, n, HD, DINP, 1, 0);
        launch_gemm(stream, Wp, bufA, HD, bufA, HD, HD, w2, HD, b2, nullptr,
                    target, HD, n, HD, HD, 0, 0);
    };
    encoder(x_Cvars, N_C,    encC_w1, encC_b1, encC_w2, encC_b2, xC);
    encoder(x_Ivars, N_I,    encI_w1, encI_b1, encI_w2, encI_b2, xI);
    encoder(x_cons,  N_CONS, encc_w1, encc_b1, encc_w2, encc_b2, xc);

    // ---- time embedding + per-layer te biases ----
    {
        TeP tp;
        for (int l = 0; l < NLAYERS; ++l) {
            tp.tw[l] = L[l].time_w; tp.tb[l] = L[l].time_b;
            tp.oib[l] = L[l].oi.b;  tp.ovb[l] = L[l].ov.b;
        }
        k_te<<<1, HD, 0, stream>>>(tsteps, time_b1, time_b2, time_w1, time_w2, tp, teI, teC);
    }

    // message_agg with matmul pushed past segment-sum:
    //   bufA = x_dst@left_w+left_b ; bufB = x_src@right_w ; S += relu(LN(gathered pre))
    //   bufT (+)= S @ fin_w + deg*fin_b
    auto msg_phase = [&](const MsgP& mp, const float* x_dst, int n_dst,
                         const float* x_src, int n_src,
                         const int* dst_ix, const int* src_ix,
                         const float* eattr, int E, const float* deg, int accum)
    {
        launch_gemm(stream, Wp, x_dst, HD, x_dst, HD, HD, mp.left_w, HD, mp.left_b, nullptr,
                    bufA, HD, n_dst, HD, HD, 0, 0);
        launch_gemm(stream, Wp, x_src, HD, x_src, HD, HD, mp.right_w, HD, nullptr, nullptr,
                    bufB, HD, n_src, HD, HD, 0, 0);
        launch_zero(stream, bufS, (long)n_dst * HD);
        k_edge<<<E, HD, 0, stream>>>(bufA, bufB, eattr, src_ix, dst_ix,
                                     mp.edge_w, mp.fln_g, mp.fln_b, bufS, E);
        launch_gemm(stream, Wp, bufS, HD, bufS, HD, HD, mp.fin_w, HD, mp.fin_b, deg,
                    bufT, HD, n_dst, HD, HD, 0, accum);
    };

    // out_block: z=[LN(agg,post), x_dst]; relu(z@out1+b) @ out2 + b -> target
    auto out_phase = [&](const OutbP& ob, const float* x_dst, int n_dst, float* target)
    {
        k_ln_rows<<<n_dst, HD, 0, stream>>>(bufT, bufA, ob.post_g, ob.post_b, n_dst, 0);
        launch_gemm(stream, Wp, bufA, HD, x_dst, HD, HD, ob.out1_w, HD, ob.out1_b, nullptr,
                    bufB, HD, n_dst, HD, 2 * HD, 1, 0);
        launch_gemm(stream, Wp, bufB, HD, bufB, HD, HD, ob.out2_w, HD, ob.out2_b, nullptr,
                    target, HD, n_dst, HD, HD, 0, 0);
    };

    // per_out: silu(LN(x)) @ w + biasVec -> target
    auto perout_phase = [&](const PerOutP& po, const float* x_in, int n, const float* biasVec,
                            float* target)
    {
        k_ln_rows<<<n, HD, 0, stream>>>(x_in, bufA, po.ln_g, po.ln_b, n, 1);
        launch_gemm(stream, Wp, bufA, HD, bufA, HD, HD, po.w, HD, biasVec, nullptr,
                    target, HD, n, HD, HD, 0, 0);
    };

    for (int l = 0; l < NLAYERS; ++l) {
        LayerP& y = L[l];
        // --- tripartite: vars -> cons ---
        msg_phase(y.triI, xc, N_CONS, xI, N_I, I_dst, I_src, I_attr, E_I, degIc, 0);
        msg_phase(y.triC, xc, N_CONS, xC, N_C, C_dst, C_src, C_attr, E_C, degCc, 1);
        out_phase(y.tri_out, xc, N_CONS, xc);
        // --- cons -> I vars (inverse edges) ---
        msg_phase(y.bI, xI, N_I, xc, N_CONS, I_src, I_dst, I_attr, E_I, degIv, 0);
        out_phase(y.bI_out, xI, N_I, bufS);                 // bufS = xI2
        perout_phase(y.oi, bufS, N_I, teI + l * HD, xI);    // xI = te + per_out(oi, xI2)
        // --- cons -> C vars ---
        msg_phase(y.bC, xC, N_C, xc, N_CONS, C_src, C_dst, C_attr, E_C, degCv, 0);
        out_phase(y.bC_out, xC, N_C, bufS);                 // bufS = xC2
        perout_phase(y.ov, bufS, N_C, teC + l * HD, xC);    // xC = te + per_out(ov, xC2)
        // --- cons self-update ---
        perout_phase(y.oc, xc, N_CONS, y.oc.b, xc);
    }

    // ---- heads ----
    // Integer head: GroupNorm(32) -> relu -> @ [128,32] + wb
    launch_zero(stream, gst, 64);
    k_gn_stats<<<dim3(32, 64), 256, 0, stream>>>(xI, N_I, gst);
    k_gn_apply_relu<<<4096, 256, 0, stream>>>(xI, bufA, gst, Iout_g, Iout_b, N_I);
    launch_gemm(stream, Wp, bufA, HD, bufA, HD, HD, Iout_w, MAXINT, Iout_wb, nullptr,
                outI, MAXINT, N_I, MAXINT, HD, 0, 0);
    // Continuous head: GroupNorm(32) -> relu -> dot with [128,1] + wb
    launch_zero(stream, gst, 64);
    k_gn_stats<<<dim3(32, 64), 256, 0, stream>>>(xC, N_C, gst);
    k_gn_apply_relu<<<4096, 256, 0, stream>>>(xC, bufA, gst, Cout_g, Cout_b, N_C);
    k_rowdot<<<N_C, HD, 0, stream>>>(bufA, Cout_w, Cout_wb, outC, N_C);

    (void)n_in; (void)out_size; (void)ws_size;
}
